// MultiHeadAttention_86784109183400
// MI455X (gfx1250) — matrix operand order repaired and checked by reading, not compiled, not run on hardware
//
#include <hip/hip_runtime.h>

typedef __bf16 bf16_t;
typedef __attribute__((ext_vector_type(16))) __bf16 v16bf;
typedef __attribute__((ext_vector_type(8)))  float  v8f;

// ---------- WMMA helpers (CDNA5 wave32 layouts, ISA 7.12.2) ----------

__device__ inline v8f wmma_bf16(v16bf a, v16bf b, v8f c) {
  // D = A(16x32 bf16) x B(32x16 bf16) + C(16x16 f32)
  return __builtin_amdgcn_wmma_f32_16x16x32_bf16(
      false, a, false, b, (short)0, c, false, false);
}

// B fragment: lane n = lane&15; elements 0..7 at K=half*8, 8..15 at K=16+half*8
__device__ inline v16bf ld_bfrag(const bf16_t* p) {
  union { v16bf v; uint4 q[2]; } u;
  u.q[0] = *(const uint4*)(p);
  u.q[1] = *(const uint4*)(p + 16);
  return u.v;
}

// A fragment from bf16 row-major [M][K]: elements 0..7 at K=half*8, 8..15 at K=16+half*8
__device__ inline v16bf ld_afrag_bf16(const bf16_t* p) {
  union { v16bf v; uint4 q[2]; } u;
  u.q[0] = *(const uint4*)(p);
  u.q[1] = *(const uint4*)(p + 16);
  return u.v;
}

// A fragment converted on the fly from fp32 row-major [M][K]
__device__ inline v16bf ld_afrag_f32(const float* p) {
  v16bf a;
#pragma unroll
  for (int e = 0; e < 8; ++e) a[e] = (bf16_t)p[e];
#pragma unroll
  for (int e = 0; e < 8; ++e) a[8 + e] = (bf16_t)p[16 + e];
  return a;
}

// ---------- constants ----------
#define NB   4
#define NSEQ 1024
#define DIM  1024
#define NH   16
#define DH   64
#define BH   (NB * NH)            // 64
#define ATTN_PER_BH ((size_t)NSEQ * NSEQ)

// Wave tile: 32(M) x 64(N)  -> acc[2][4], 8 WMMA per K-step of 32.

// ---------- kernel 0: transpose/convert weights to bf16 [N][K]; Er -> bf16 ----------
__global__ __launch_bounds__(256) void prep_kernel(
    const float* __restrict__ Wqkv, const float* __restrict__ Wout,
    const float* __restrict__ Er,
    bf16_t* __restrict__ WqkvT, bf16_t* __restrict__ WoutT, bf16_t* __restrict__ ErB) {
  const size_t N1 = (size_t)3 * DIM * DIM;   // WqkvT elems
  const size_t N2 = (size_t)DIM * DIM;       // WoutT elems
  const size_t N3 = (size_t)NSEQ * DH;       // Er elems
  for (size_t idx = (size_t)blockIdx.x * 256 + threadIdx.x; idx < N1 + N2 + N3;
       idx += (size_t)gridDim.x * 256) {
    if (idx < N1) {
      size_t n = idx >> 10, k = idx & 1023;          // [n][k], n<3072
      WqkvT[idx] = (bf16_t)Wqkv[k * (3 * DIM) + n];
    } else if (idx < N1 + N2) {
      size_t t = idx - N1;
      size_t n = t >> 10, k = t & 1023;
      WoutT[t] = (bf16_t)Wout[k * DIM + n];
    } else {
      size_t t = idx - N1 - N2;
      ErB[t] = (bf16_t)Er[t];                        // already [n][d] = [N][K]
    }
  }
}

// ---------- kernel 1: QKV GEMM (M=4096,N=3072,K=1024), scatter q/k/vT bf16 ----------
// Block tile 128x128 (8 waves as 4M x 2N). N-block never crosses a 1024-col section,
// wave N-tile (64) never crosses a head, M-block (128) never crosses a batch ->
// section/head/batch are wave-uniform scalars.
__global__ __launch_bounds__(256) void qkv_kernel(
    const float* __restrict__ x, const bf16_t* __restrict__ WqkvT,
    bf16_t* __restrict__ qb, bf16_t* __restrict__ kb, bf16_t* __restrict__ vt) {
  const int lane = threadIdx.x & 31, wave = threadIdx.x >> 5;
  const int wm = wave >> 1, wn = wave & 1;
  const int m0 = blockIdx.y * 128 + wm * 32;
  const int n0 = blockIdx.x * 128 + wn * 64;
  const int rr = lane & 15, half = lane >> 4;
  v8f acc[2][4] = {};
  for (int k0 = 0; k0 < DIM; k0 += 32) {
    v16bf a0 = ld_afrag_f32(x + (size_t)(m0 + rr) * DIM + k0 + half * 8);
    v16bf a1 = ld_afrag_f32(x + (size_t)(m0 + 16 + rr) * DIM + k0 + half * 8);
    v16bf b[4];
#pragma unroll
    for (int fj = 0; fj < 4; ++fj)
      b[fj] = ld_bfrag(WqkvT + (size_t)(n0 + fj * 16 + rr) * DIM + k0 + half * 8);
#pragma unroll
    for (int fj = 0; fj < 4; ++fj) {
      acc[0][fj] = wmma_bf16(a0, b[fj], acc[0][fj]);
      acc[1][fj] = wmma_bf16(a1, b[fj], acc[1][fj]);
    }
  }
  // wave-uniform decode
  const int sec = n0 >> 10;            // 0=q, 1=k, 2=v
  const int hh  = (n0 & 1023) >> 6;    // head (wave-uniform: wave N-tile == one head)
  const int bb  = m0 >> 10;            // batch (wave-uniform)
  const int tok0 = m0 & 1023;
  const int bh = bb * NH + hh;
  if (sec == 2) {
    // v transposed: vt[bh][dd][tok]
    bf16_t* base = vt + ((size_t)bh * DH) * NSEQ;
#pragma unroll
    for (int fi = 0; fi < 2; ++fi)
#pragma unroll
      for (int fj = 0; fj < 4; ++fj)
#pragma unroll
        for (int r = 0; r < 8; ++r) {
          int tok = tok0 + fi * 16 + half * 8 + r;
          int dd  = fj * 16 + rr;
          base[(size_t)dd * NSEQ + tok] = (bf16_t)acc[fi][fj][r];
        }
  } else {
    bf16_t* base = (sec == 0 ? qb : kb) + ((size_t)bh * NSEQ + tok0) * DH;
#pragma unroll
    for (int fi = 0; fi < 2; ++fi)
#pragma unroll
      for (int fj = 0; fj < 4; ++fj)
#pragma unroll
        for (int r = 0; r < 8; ++r) {
          int trow = fi * 16 + half * 8 + r;
          int dd   = fj * 16 + rr;
          base[(size_t)trow * DH + dd] = (bf16_t)acc[fi][fj][r];
        }
  }
}

// ---------- kernel 2: dots = scale * q k^T  (per bh, M=N=1024, K=64) ----------
__global__ __launch_bounds__(256) void qk_kernel(
    const bf16_t* __restrict__ qb, const bf16_t* __restrict__ kb,
    float* __restrict__ attn) {
  const int bh = blockIdx.z;
  const int lane = threadIdx.x & 31, wave = threadIdx.x >> 5;
  const int wm = wave >> 1, wn = wave & 1;
  const int m0 = blockIdx.y * 128 + wm * 32;
  const int j0 = blockIdx.x * 128 + wn * 64;
  const int rr = lane & 15, half = lane >> 4;
  const bf16_t* qp = qb + (size_t)bh * NSEQ * DH;
  const bf16_t* kp = kb + (size_t)bh * NSEQ * DH;
  v8f acc[2][4] = {};
#pragma unroll
  for (int k0 = 0; k0 < DH; k0 += 32) {
    v16bf a0 = ld_afrag_bf16(qp + (size_t)(m0 + rr) * DH + k0 + half * 8);
    v16bf a1 = ld_afrag_bf16(qp + (size_t)(m0 + 16 + rr) * DH + k0 + half * 8);
    v16bf b[4];
#pragma unroll
    for (int fj = 0; fj < 4; ++fj)
      b[fj] = ld_bfrag(kp + (size_t)(j0 + fj * 16 + rr) * DH + k0 + half * 8);
#pragma unroll
    for (int fj = 0; fj < 4; ++fj) {
      acc[0][fj] = wmma_bf16(a0, b[fj], acc[0][fj]);
      acc[1][fj] = wmma_bf16(a1, b[fj], acc[1][fj]);
    }
  }
  float* dst = attn + (size_t)bh * ATTN_PER_BH;
  const float scale = 0.125f;  // 64^-0.5
#pragma unroll
  for (int fi = 0; fi < 2; ++fi)
#pragma unroll
    for (int fj = 0; fj < 4; ++fj)
#pragma unroll
      for (int r = 0; r < 8; ++r) {
        int m = m0 + fi * 16 + half * 8 + r;
        int j = j0 + fj * 16 + rr;
        dst[(size_t)m * NSEQ + j] = scale * acc[fi][fj][r];
      }
}

// ---------- kernel 3: R = q Er^T, scatter srel[i, i+m-1023] += R[i,m] (i+m>=1023) ----------
__global__ __launch_bounds__(256) void qe_kernel(
    const bf16_t* __restrict__ qb, const bf16_t* __restrict__ ErB,
    float* __restrict__ attn) {
  const int bh = blockIdx.z;
  const int lane = threadIdx.x & 31, wave = threadIdx.x >> 5;
  const int wm = wave >> 1, wn = wave & 1;
  const int i0  = blockIdx.y * 128 + wm * 32;
  const int mm0 = blockIdx.x * 128 + wn * 64;
  if (i0 + 31 + mm0 + 63 < NSEQ - 1) return;  // whole 32x64 wave tile masked (wave-uniform)
  const int rr = lane & 15, half = lane >> 4;
  const bf16_t* qp = qb + (size_t)bh * NSEQ * DH;
  v8f acc[2][4] = {};
#pragma unroll
  for (int k0 = 0; k0 < DH; k0 += 32) {
    v16bf a0 = ld_afrag_bf16(qp + (size_t)(i0 + rr) * DH + k0 + half * 8);
    v16bf a1 = ld_afrag_bf16(qp + (size_t)(i0 + 16 + rr) * DH + k0 + half * 8);
    v16bf b[4];
#pragma unroll
    for (int fj = 0; fj < 4; ++fj)
      b[fj] = ld_bfrag(ErB + (size_t)(mm0 + fj * 16 + rr) * DH + k0 + half * 8);
#pragma unroll
    for (int fj = 0; fj < 4; ++fj) {
      acc[0][fj] = wmma_bf16(a0, b[fj], acc[0][fj]);
      acc[1][fj] = wmma_bf16(a1, b[fj], acc[1][fj]);
    }
  }
  float* dst = attn + (size_t)bh * ATTN_PER_BH;
#pragma unroll
  for (int fi = 0; fi < 2; ++fi)
#pragma unroll
    for (int fj = 0; fj < 4; ++fj)
#pragma unroll
      for (int r = 0; r < 8; ++r) {
        int i = i0 + fi * 16 + half * 8 + r;
        int m = mm0 + fj * 16 + rr;
        int j = i + m - (NSEQ - 1);
        if (j >= 0) {  // j <= i < NSEQ guaranteed
          size_t idx = (size_t)i * NSEQ + j;
          dst[idx] += acc[fi][fj][r];
        }
      }
}

// ---------- kernel 4: row softmax in place (one block per row of 1024) ----------
__global__ __launch_bounds__(256) void softmax_kernel(float* __restrict__ attn) {
  float* p = attn + (size_t)blockIdx.x * NSEQ;
  const int t = threadIdx.x;
  float a = p[t], b = p[t + 256], c = p[t + 512], d = p[t + 768];
  __shared__ float red[256];
  red[t] = fmaxf(fmaxf(a, b), fmaxf(c, d));
  __syncthreads();
  for (int s = 128; s > 0; s >>= 1) {
    if (t < s) red[t] = fmaxf(red[t], red[t + s]);
    __syncthreads();
  }
  float mx = red[0];
  __syncthreads();
  float e0 = __expf(a - mx), e1 = __expf(b - mx), e2 = __expf(c - mx), e3 = __expf(d - mx);
  red[t] = e0 + e1 + e2 + e3;
  __syncthreads();
  for (int s = 128; s > 0; s >>= 1) {
    if (t < s) red[t] += red[t + s];
    __syncthreads();
  }
  float inv = 1.0f / red[0];
  p[t] = e0 * inv; p[t + 256] = e1 * inv; p[t + 512] = e2 * inv; p[t + 768] = e3 * inv;
}

// ---------- kernel 5: ctx = attn @ v  (per bh, M=1024, N=64, K=1024) ----------
// 8 waves all along M (block tile 256 x 64); each wave's 32x64 tile spans full N.
__global__ __launch_bounds__(256) void av_kernel(
    const float* __restrict__ attn, const bf16_t* __restrict__ vt,
    bf16_t* __restrict__ ctx) {
  const int bh = blockIdx.z;
  const int lane = threadIdx.x & 31, wave = threadIdx.x >> 5;
  const int m0 = blockIdx.y * 256 + wave * 32;
  const int rr = lane & 15, half = lane >> 4;
  const float*  ap = attn + (size_t)bh * ATTN_PER_BH;
  const bf16_t* vp = vt + (size_t)bh * DH * NSEQ;   // [d][n]
  v8f acc[2][4] = {};
  for (int k0 = 0; k0 < NSEQ; k0 += 32) {
    v16bf a0 = ld_afrag_f32(ap + (size_t)(m0 + rr) * NSEQ + k0 + half * 8);
    v16bf a1 = ld_afrag_f32(ap + (size_t)(m0 + 16 + rr) * NSEQ + k0 + half * 8);
    v16bf b[4];
#pragma unroll
    for (int fj = 0; fj < 4; ++fj)
      b[fj] = ld_bfrag(vp + (size_t)(fj * 16 + rr) * NSEQ + k0 + half * 8);
#pragma unroll
    for (int fj = 0; fj < 4; ++fj) {
      acc[0][fj] = wmma_bf16(a0, b[fj], acc[0][fj]);
      acc[1][fj] = wmma_bf16(a1, b[fj], acc[1][fj]);
    }
  }
  const int bb = bh >> 4, hh = bh & 15;
  // ctx layout [b][n][h*64+d]  (== transpose(0,2,1,3).reshape)
  bf16_t* base = ctx + ((size_t)bb * NSEQ + m0) * DIM + hh * DH;
#pragma unroll
  for (int fi = 0; fi < 2; ++fi)
#pragma unroll
    for (int fj = 0; fj < 4; ++fj)
#pragma unroll
      for (int r = 0; r < 8; ++r) {
        int trow = fi * 16 + half * 8 + r;
        int dd   = fj * 16 + rr;
        base[(size_t)trow * DIM + dd] = (bf16_t)acc[fi][fj][r];
      }
}

// ---------- kernel 6: out = ctx @ Wout + bout  (M=4096, N=1024, K=1024) ----------
__global__ __launch_bounds__(256) void outproj_kernel(
    const bf16_t* __restrict__ ctx, const bf16_t* __restrict__ WoutT,
    const float* __restrict__ bout, float* __restrict__ out) {
  const int lane = threadIdx.x & 31, wave = threadIdx.x >> 5;
  const int wm = wave >> 1, wn = wave & 1;
  const int m0 = blockIdx.y * 128 + wm * 32;
  const int n0 = blockIdx.x * 128 + wn * 64;
  const int rr = lane & 15, half = lane >> 4;
  v8f acc[2][4] = {};
  for (int k0 = 0; k0 < DIM; k0 += 32) {
    v16bf a0 = ld_afrag_bf16(ctx + (size_t)(m0 + rr) * DIM + k0 + half * 8);
    v16bf a1 = ld_afrag_bf16(ctx + (size_t)(m0 + 16 + rr) * DIM + k0 + half * 8);
    v16bf b[4];
#pragma unroll
    for (int fj = 0; fj < 4; ++fj)
      b[fj] = ld_bfrag(WoutT + (size_t)(n0 + fj * 16 + rr) * DIM + k0 + half * 8);
#pragma unroll
    for (int fj = 0; fj < 4; ++fj) {
      acc[0][fj] = wmma_bf16(a0, b[fj], acc[0][fj]);
      acc[1][fj] = wmma_bf16(a1, b[fj], acc[1][fj]);
    }
  }
#pragma unroll
  for (int fi = 0; fi < 2; ++fi)
#pragma unroll
    for (int fj = 0; fj < 4; ++fj)
#pragma unroll
      for (int r = 0; r < 8; ++r) {
        int m  = m0 + fi * 16 + half * 8 + r;
        int nn = n0 + fj * 16 + rr;
        out[(size_t)m * DIM + nn] = acc[fi][fj][r] + bout[nn];
      }
}

// ---------- host launch ----------
extern "C" void kernel_launch(void* const* d_in, const int* in_sizes, int n_in,
                              void* d_out, int out_size, void* d_ws, size_t ws_size,
                              hipStream_t stream) {
  const float* x    = (const float*)d_in[0];
  const float* Wqkv = (const float*)d_in[1];
  const float* Er   = (const float*)d_in[2];
  const float* Wout = (const float*)d_in[3];
  const float* bout = (const float*)d_in[4];

  float* out  = (float*)d_out;
  float* attn = out + (size_t)NB * NSEQ * DIM;   // 4M floats offset

  char* ws = (char*)d_ws;
  bf16_t* WqkvT = (bf16_t*)ws;  ws += (size_t)3 * DIM * DIM * sizeof(bf16_t);   // 6 MB
  bf16_t* WoutT = (bf16_t*)ws;  ws += (size_t)DIM * DIM * sizeof(bf16_t);       // 2 MB
  bf16_t* ErB   = (bf16_t*)ws;  ws += (size_t)NSEQ * DH * sizeof(bf16_t);       // 128 KB
  bf16_t* qb    = (bf16_t*)ws;  ws += (size_t)BH * NSEQ * DH * sizeof(bf16_t);  // 8 MB
  bf16_t* kb    = (bf16_t*)ws;  ws += (size_t)BH * NSEQ * DH * sizeof(bf16_t);  // 8 MB
  bf16_t* vt    = (bf16_t*)ws;  ws += (size_t)BH * DH * NSEQ * sizeof(bf16_t);  // 8 MB
  bf16_t* ctx   = (bf16_t*)ws;  ws += (size_t)NB * NSEQ * DIM * sizeof(bf16_t); // 8 MB

  prep_kernel<<<4096, 256, 0, stream>>>(Wqkv, Wout, Er, WqkvT, WoutT, ErB);
  qkv_kernel<<<dim3(24, 32), 256, 0, stream>>>(x, WqkvT, qb, kb, vt);
  qk_kernel<<<dim3(8, 8, BH), 256, 0, stream>>>(qb, kb, attn);
  qe_kernel<<<dim3(8, 8, BH), 256, 0, stream>>>(qb, ErB, attn);
  softmax_kernel<<<BH * NSEQ, 256, 0, stream>>>(attn);
  av_kernel<<<dim3(1, 4, BH), 256, 0, stream>>>(attn, vt, ctx);
  outproj_kernel<<<dim3(8, 32), 256, 0, stream>>>(ctx, WoutT, bout, out);
}
